// GCN_16999480558417
// MI455X (gfx1250) — compile-verified, matrix operand order checked by reading, not run on hardware
//
#include <hip/hip_runtime.h>
#include <hip/hip_bf16.h>
#include <math.h>

// ---------------------------------------------------------------------------
// GCN (2x dense-adj graph conv, 4096x1024) + log_softmax/l2norm + GRU scan.
// bf16 WMMA (v_wmma_f32_16x16x32_bf16, f32 accum) for all GEMMs; f32 for the
// softmax/l2norm and GRU nonlinearity. Serial GRU reads a packed bf16 W_hh
// ([k][j][r,z,n,pad]) that stays L2-resident: one b64 load per (k,thread).
// GEMM: 128x128 block tile, BK=32, register-pipelined global->LDS staging
// (uint4 global loads overlap the WMMA issue of the previous tile).
// ---------------------------------------------------------------------------

typedef __attribute__((ext_vector_type(16))) __bf16 v16bf;
typedef __attribute__((ext_vector_type(8)))  float  v8f;

__device__ __forceinline__ v8f wmma_bf16(v16bf a, v16bf b, v8f c) {
#if defined(__gfx1250__)
  // 8 args: (neg_a, A, neg_b, B, c_mod, C, reuse_a, reuse_b)
  return __builtin_amdgcn_wmma_f32_16x16x32_bf16(false, a, false, b, (short)0, c,
                                                 false, false);
#else
  (void)a; (void)b;
  return c;
#endif
}

union Pk4 { uint2 u; __bf16 e[4]; };

// ------------- vectorized f32 -> bf16 cast: float4 in, 4x bf16 out ---------
__global__ void k_cast4_bf16(const float4* __restrict__ s, uint2* __restrict__ d, int n4) {
  int i = blockIdx.x * blockDim.x + threadIdx.x;
  int st = gridDim.x * blockDim.x;
  for (; i < n4; i += st) {
    float4 f = s[i];
    Pk4 p;
    p.e[0] = (__bf16)f.x; p.e[1] = (__bf16)f.y;
    p.e[2] = (__bf16)f.z; p.e[3] = (__bf16)f.w;
    d[i] = p.u;
  }
}

// ------------- tiled transpose + cast: src[R][C] f32 -> dst[C][R] bf16 ------
__global__ void __launch_bounds__(256)
k_tcast_bf16(const float* __restrict__ s, __bf16* __restrict__ d, int R, int C) {
  __shared__ float tile[32][33];
  const int c0 = blockIdx.x * 32, r0 = blockIdx.y * 32;
  for (int i = threadIdx.y; i < 32; i += 8)
    tile[i][threadIdx.x] = s[(size_t)(r0 + i) * C + c0 + threadIdx.x];
  __syncthreads();
  for (int i = threadIdx.y; i < 32; i += 8)
    d[(size_t)(c0 + i) * R + r0 + threadIdx.x] = (__bf16)tile[threadIdx.x][i];
}

// ------ pack W_hh[3E,E] -> whh4[k][j][4] = {r,z,n,0} bf16 (b64 per (k,j)) ---
__global__ void __launch_bounds__(256)
k_pack_whh(const float* __restrict__ whh, __bf16* __restrict__ d) {
  int idx = blockIdx.x * blockDim.x + threadIdx.x;  // 0 .. 1024*1024-1
  int k = idx >> 10, j = idx & 1023;
  Pk4 p;
  p.e[0] = (__bf16)whh[(size_t)(j)        * 1024 + k];
  p.e[1] = (__bf16)whh[(size_t)(j + 1024) * 1024 + k];
  p.e[2] = (__bf16)whh[(size_t)(j + 2048) * 1024 + k];
  p.e[3] = (__bf16)0.f;
  *(uint2*)&d[(size_t)idx * 4] = p.u;
}

// ---------------------------------------------------------------------------
// bf16 WMMA GEMM: C[M,N] = A[M,K] @ B[K,N] (+bias[N]) (+ReLU)
// 256 threads = 8 waves in a 2(M) x 4(N) grid; each wave: 64x32 patch =
// 4x2 WMMA tiles. Global staging: uint4 (b128) loads into registers,
// pipelined one K-tile ahead of the WMMAs; LDS commit via b32 (A) / b16 (B) stores into
// padded [m][k]/[n][k] layouts so fragment gathers are ds_load_2addr_b32.
// All dims are multiples of 128/32 -> no bounds checks.
// ---------------------------------------------------------------------------
template <bool RELU>
__global__ void __launch_bounds__(256)
k_gemm_bf16(int M, int N, int K,
            const __bf16* __restrict__ A, const __bf16* __restrict__ B,
            const float* __restrict__ bias,
            float* __restrict__ outF, __bf16* __restrict__ outH) {
  constexpr int BM = 128, BN = 128, BK = 32;
  constexpr int SA = BK + 2;   // padded LDS strides (halves): conflict-free
  constexpr int SB = BK + 2;
  __shared__ __bf16 As[BM * SA];      // [m][k]
  __shared__ __bf16 Bs[BN * SB];      // transposed: [n][k]

  const int tid  = threadIdx.x;
  const int lane = tid & 31;
  const int wave = tid >> 5;
  const int wm   = wave >> 2;         // 0..1 : 64 rows each
  const int wn   = wave & 3;          // 0..3 : 32 cols each
  const int m0   = blockIdx.y * BM;
  const int n0   = blockIdx.x * BN;

  // per-thread staging coordinates: 2 x 16B chunks of A and of B
  int ar[2], ac[2], bk[2], bc[2];
  #pragma unroll
  for (int p = 0; p < 2; ++p) {
    int ia = tid + p * 256;           // 0..511
    ar[p] = ia >> 2;  ac[p] = (ia & 3) * 8;     // A: 128 rows x 32 halves
    bk[p] = ia >> 4;  bc[p] = (ia & 15) * 8;    // B: 32 rows x 128 halves
  }

  union Q  { uint4 q; uint w[4]; __bf16 e[8]; };
  union CF { v8f v; float e[8]; };
  union AF { v16bf v; __bf16 e[16]; };

  Q ra[2], rb[2];

  auto load_tiles = [&](int k0) {
    #pragma unroll
    for (int p = 0; p < 2; ++p) {
      ra[p].q = *(const uint4*)&A[(size_t)(m0 + ar[p]) * K + k0 + ac[p]];
      rb[p].q = *(const uint4*)&B[(size_t)(k0 + bk[p]) * N + n0 + bc[p]];
    }
  };
  auto store_tiles = [&]() {
    #pragma unroll
    for (int p = 0; p < 2; ++p) {
      uint* da = (uint*)&As[ar[p] * SA + ac[p]];     // 4B-aligned (SA even)
      #pragma unroll
      for (int q = 0; q < 4; ++q) da[q] = ra[p].w[q];
      #pragma unroll
      for (int q = 0; q < 8; ++q) Bs[(bc[p] + q) * SB + bk[p]] = rb[p].e[q];
    }
  };

  CF acc[4][2];
  #pragma unroll
  for (int i = 0; i < 4; ++i)
    #pragma unroll
    for (int j = 0; j < 2; ++j)
      #pragma unroll
      for (int r = 0; r < 8; ++r) acc[i][j].e[r] = 0.f;

  const int mrow = wm * 64 + (lane & 15);
  const int hi8  = (lane >> 4) * 8;   // K sub-offset AND C/D row sub-offset
  const int ncol = wn * 32 + (lane & 15);

  load_tiles(0);
  store_tiles();
  __syncthreads();

  for (int k0 = 0;;) {
    // ---- gather fragments per ISA 16-bit A/B layout (16x32 / 32x16) ----
    AF afr[4], bfr[2];
    #pragma unroll
    for (int i = 0; i < 4; ++i) {
      const __bf16* ap = &As[(mrow + i * 16) * SA];
      #pragma unroll
      for (int j = 0; j < 16; ++j) {
        int v2 = j >> 1;
        int kk = ((v2 < 4) ? (2 * v2) : (16 + 2 * (v2 - 4))) + (j & 1) + hi8;
        afr[i].e[j] = ap[kk];
      }
    }
    #pragma unroll
    for (int jj = 0; jj < 2; ++jj) {
      const __bf16* bp = &Bs[(ncol + jj * 16) * SB];
      #pragma unroll
      for (int j = 0; j < 16; ++j) {
        int v2 = j >> 1;
        int kk = ((v2 < 4) ? (2 * v2) : (16 + 2 * (v2 - 4))) + (j & 1) + hi8;
        bfr[jj].e[j] = bp[kk];
      }
    }

    const int kn = k0 + BK;
    if (kn < K) load_tiles(kn);       // issue next-tile b128 loads early:
                                      // they overlap the WMMA issue below
    #pragma unroll
    for (int i = 0; i < 4; ++i)
      #pragma unroll
      for (int jj = 0; jj < 2; ++jj)
        acc[i][jj].v = wmma_bf16(afr[i].v, bfr[jj].v, acc[i][jj].v);

    k0 = kn;
    if (k0 >= K) break;
    __syncthreads();                  // everyone done reading LDS
    store_tiles();                    // commit prefetched tile
    __syncthreads();
  }

  // ---- epilogue: bias / ReLU / f32 and/or bf16 stores ----
  #pragma unroll
  for (int i = 0; i < 4; ++i) {
    #pragma unroll
    for (int jj = 0; jj < 2; ++jj) {
      #pragma unroll
      for (int r = 0; r < 8; ++r) {
        int m = m0 + wm * 64 + i * 16 + r + hi8;
        int n = n0 + wn * 32 + jj * 16 + (lane & 15);
        float v = acc[i][jj].e[r];
        if (bias) v += bias[n];
        if (RELU) v = fmaxf(v, 0.f);
        size_t o = (size_t)m * N + n;
        if (outF) outF[o] = v;
        if (outH) outH[o] = (__bf16)v;
      }
    }
  }
}

// ------------- row-wise log_softmax + l2norm, 1 block / row (C=1024) -------
__global__ void __launch_bounds__(256)
k_logsoftmax_l2(const float* __restrict__ x, __bf16* __restrict__ oh, int C) {
  const int row = blockIdx.x, tid = threadIdx.x;
  const float* xr = x + (size_t)row * C;
  __shared__ float red[256];
  float4 f = ((const float4*)xr)[tid];          // elements 4t..4t+3
  float v[4] = {f.x, f.y, f.z, f.w};
  float mx = fmaxf(fmaxf(v[0], v[1]), fmaxf(v[2], v[3]));
  red[tid] = mx; __syncthreads();
  for (int s = 128; s > 0; s >>= 1) { if (tid < s) red[tid] = fmaxf(red[tid], red[tid + s]); __syncthreads(); }
  mx = red[0]; __syncthreads();

  float se = 0.f;
  #pragma unroll
  for (int i = 0; i < 4; ++i) se += __expf(v[i] - mx);
  red[tid] = se; __syncthreads();
  for (int s = 128; s > 0; s >>= 1) { if (tid < s) red[tid] += red[tid + s]; __syncthreads(); }
  const float lse = mx + __logf(red[0]); __syncthreads();

  float ss = 0.f;
  #pragma unroll
  for (int i = 0; i < 4; ++i) { v[i] -= lse; ss += v[i] * v[i]; }
  red[tid] = ss; __syncthreads();
  for (int s = 128; s > 0; s >>= 1) { if (tid < s) red[tid] += red[tid + s]; __syncthreads(); }
  const float rn = rsqrtf(red[0]);

  Pk4 p;
  #pragma unroll
  for (int i = 0; i < 4; ++i) p.e[i] = (__bf16)(v[i] * rn);
  ((uint2*)(oh + (size_t)row * C))[tid] = p.u;
}

// ---------------------------------------------------------------------------
// GRU scan: strictly serial over T=4096 steps. One persistent 1024-thread
// workgroup (32 waves); h lives in LDS; thread j owns hidden unit j. The
// three recurrent dots read the packed bf16 whh4[k][j][r,z,n,_] with a single
// coalesced b64 load per k (8.4MB/step, L2-resident).
// ---------------------------------------------------------------------------
__global__ void __launch_bounds__(1024)
k_gru(const float* __restrict__ gi, const __bf16* __restrict__ whh4,
      const float* __restrict__ b_hh, float* __restrict__ out, int T) {
  const int j = threadIdx.x;                 // 0..1023
  __shared__ float hs[1024];
  hs[j] = 0.f;
  __syncthreads();
  const float bh_r = b_hh[j], bh_z = b_hh[j + 1024], bh_n = b_hh[j + 2048];

  for (int t = 0; t < T; ++t) {
    const float* git = gi + (size_t)t * 3072;
    float gr = bh_r, gz = bh_z, gn = bh_n;
    for (int k = 0; k < 1024; ++k) {
      const float hk = hs[k];
      Pk4 p;
      p.u = *(const uint2*)&whh4[((size_t)k * 1024 + j) * 4];
      gr += hk * (float)p.e[0];
      gz += hk * (float)p.e[1];
      gn += hk * (float)p.e[2];
    }
    const float r = 1.f / (1.f + __expf(-(git[j] + gr)));
    const float z = 1.f / (1.f + __expf(-(git[j + 1024] + gz)));
    const float n = tanhf(git[j + 2048] + r * gn);
    const float hnew = (1.f - z) * n + z * hs[j];
    out[(size_t)t * 1024 + j] = hnew;
    __syncthreads();
    hs[j] = hnew;
    __syncthreads();
  }
}

// ---------------------------------------------------------------------------
extern "C" void kernel_launch(void* const* d_in, const int* in_sizes, int n_in,
                              void* d_out, int out_size, void* d_ws, size_t ws_size,
                              hipStream_t stream) {
  (void)in_sizes; (void)n_in; (void)out_size; (void)ws_size;
  const float* word  = (const float*)d_in[0];
  const float* adj   = (const float*)d_in[1];
  // d_in[2]=W_fc, d_in[3]=b_fc: dead branch in the reference, skipped.
  const float* gc1_W = (const float*)d_in[4];
  const float* gc1_b = (const float*)d_in[5];
  const float* gc2_W = (const float*)d_in[6];
  const float* gc2_b = (const float*)d_in[7];
  const float* W_ih  = (const float*)d_in[8];
  const float* W_hh  = (const float*)d_in[9];
  const float* b_ih  = (const float*)d_in[10];
  const float* b_hh  = (const float*)d_in[11];
  float* out = (float*)d_out;
  char* ws = (char*)d_ws;

  // workspace layout (bytes)
  constexpr size_t OFF_ADJH = 0;                         // adj bf16   33.55 MB
  constexpr size_t OFF_H0   = OFF_ADJH + 33554432ull;    // 4096x1024 bf16 (word_h / xn_h)
  constexpr size_t OFF_H1   = OFF_H0   + 8388608ull;     // 4096x1024 bf16 (t1 / t2)
  constexpr size_t OFF_H2   = OFF_H1   + 8388608ull;     // 4096x1024 bf16 (x1)
  constexpr size_t OFF_WH   = OFF_H2   + 8388608ull;     // weight bf16 (up to 1024x3072)
  constexpr size_t OFF_XF   = OFF_WH   + 6291456ull;     // x2 f32   16.78 MB
  constexpr size_t OFF_GI   = OFF_XF   + 16777216ull;    // gi f32   50.33 MB
  constexpr size_t OFF_WHH4 = OFF_GI   + 50331648ull;    // packed W_hh bf16 8.39 MB

  __bf16* adjh = (__bf16*)(ws + OFF_ADJH);
  __bf16* h0   = (__bf16*)(ws + OFF_H0);
  __bf16* h1   = (__bf16*)(ws + OFF_H1);
  __bf16* h2   = (__bf16*)(ws + OFF_H2);
  __bf16* wh   = (__bf16*)(ws + OFF_WH);
  float*  xf   = (float*) (ws + OFF_XF);
  float*  gi   = (float*) (ws + OFF_GI);
  __bf16* whh4 = (__bf16*)(ws + OFF_WHH4);

  const dim3 b256(256);

  // one-time bf16 casts (float4 -> packed 4x bf16)
  k_cast4_bf16<<<dim3(4096), b256, 0, stream>>>((const float4*)adj,  (uint2*)adjh, 4096 * 4096 / 4);
  k_cast4_bf16<<<dim3(4096), b256, 0, stream>>>((const float4*)word, (uint2*)h0,   4096 * 1024 / 4);
  k_cast4_bf16<<<dim3(1024), b256, 0, stream>>>((const float4*)gc1_W, (uint2*)wh,  1024 * 1024 / 4);

  // t1 = word @ gc1_W                       -> h1 (bf16)
  k_gemm_bf16<false><<<dim3(8, 32), b256, 0, stream>>>(4096, 1024, 1024, h0, wh, nullptr, nullptr, h1);
  // x1 = relu(adj @ t1 + gc1_b)             -> h2 (bf16)
  k_gemm_bf16<true ><<<dim3(8, 32), b256, 0, stream>>>(4096, 1024, 4096, adjh, h1, gc1_b, nullptr, h2);
  k_cast4_bf16<<<dim3(1024), b256, 0, stream>>>((const float4*)gc2_W, (uint2*)wh, 1024 * 1024 / 4);
  // t2 = x1 @ gc2_W                         -> h1 (bf16)
  k_gemm_bf16<false><<<dim3(8, 32), b256, 0, stream>>>(4096, 1024, 1024, h2, wh, nullptr, nullptr, h1);
  // x2 = adj @ t2 + gc2_b                   -> xf (f32)
  k_gemm_bf16<false><<<dim3(8, 32), b256, 0, stream>>>(4096, 1024, 4096, adjh, h1, gc2_b, xf, nullptr);

  // xn = l2norm(log_softmax(x2))            -> h0 (bf16)
  k_logsoftmax_l2<<<dim3(4096), b256, 0, stream>>>(xf, h0, 1024);

  // W_ih^T (f32 [3072,1024] -> bf16 [1024,3072]) and gi = xn @ W_ih^T + b_ih
  k_tcast_bf16<<<dim3(32, 96), dim3(32, 8), 0, stream>>>(W_ih, wh, 3072, 1024);
  k_gemm_bf16<false><<<dim3(24, 32), b256, 0, stream>>>(4096, 3072, 1024, h0, wh, b_ih, gi, nullptr);

  // packed W_hh and the serial GRU scan
  k_pack_whh<<<dim3(4096), b256, 0, stream>>>(W_hh, whh4);
  k_gru<<<dim3(1), dim3(1024), 0, stream>>>(gi, whh4, b_hh, out, 4096);
}
